// nmt_model_51642686767288
// MI455X (gfx1250) — matrix operand order repaired and checked by reading, not compiled, not run on hardware
//
#include <hip/hip_runtime.h>

// ---------------------------------------------------------------------------
// Model dims
// ---------------------------------------------------------------------------
#define BATCH 64
#define SRCLEN 50
#define TRGLEN 30
#define VOCAB 32000
#define ENC_EMB 256
#define ENC_H 256
#define DEC_EMB 256
#define DEC_H 512
#define ATT 512
#define BS (BATCH * SRCLEN)   // 3200
#define BT (BATCH * TRGLEN)   // 1920

typedef __attribute__((ext_vector_type(8)))  __bf16 v8bf;
typedef __attribute__((ext_vector_type(16))) __bf16 v16bf;
typedef __attribute__((ext_vector_type(8)))  float  v8f;

// ---------------------------------------------------------------------------
// bf16 WMMA GEMM with 2x2 register blocking (one wave -> 32x32 output tile):
//   C[M,N] = A[M,K] @ W[N,K]^T (+bias1[n]) (+bias2[n]) (+addend[m,n])
// fp32 accumulate. A row-major bf16 (lda), W row-major bf16 [N,K] (ldw), so
// the B operand is W^T and each lane's 16 K values of a B column (K 8h..8h+7, 16+8h..23+8h)
// are two 16-byte reads from the same W row. CDNA5 wave32 WMMA VGPR layouts per
// ISA 7.12.2. Per K-step: 8x global_load_b128 feed 4x v_wmma (2 loads/WMMA).
// Optional store-permutation: output row r -> (r % pb)*pt + r/pb  (t,b -> b,t)
// All M,N used here are multiples of 32; K multiples of 32.
// ---------------------------------------------------------------------------
__global__ void __launch_bounds__(256)
gemm_bf16_wmma(const __bf16* __restrict__ A, int lda,
               const __bf16* __restrict__ W, int ldw,
               float* __restrict__ C, int ldc,
               const float* __restrict__ bias1,
               const float* __restrict__ bias2,
               const float* __restrict__ addend, int ldadd,
               int M, int N, int K,
               int permute, int pb, int pt)
{
    const int wave = threadIdx.x >> 5;
    const int lane = threadIdx.x & 31;
    const int Mt = M >> 5, Nt = N >> 5;          // 32x32 macro-tiles
    const int tile = blockIdx.x * 8 + wave;
    if (tile >= Mt * Nt) return;                 // wave-uniform: EXEC all-ones inside
    const int mt = tile / Nt;
    const int nt = tile % Nt;

    // A operand rows: 16x32 bf16 tile. lane&15 = M row; lanes>=16 take K base +8.
    const int arow  = (mt << 5) + (lane & 15);
    const int abase = (lane >> 4) << 3;          // 0 or 8
    const __bf16* __restrict__ Ap0 = A + (size_t)arow * lda + abase;
    const __bf16* __restrict__ Ap1 = Ap0 + (size_t)16 * lda;

    // B operand cols from W[N,K]: lane&15 = N col; lanes>=16 take K base +8.
    const int wrow  = (nt << 5) + (lane & 15);
    const int bbase = (lane >> 4) << 3;          // 0 or 8
    const __bf16* __restrict__ Wp0 = W + (size_t)wrow * ldw + bbase;
    const __bf16* __restrict__ Wp1 = Wp0 + (size_t)16 * ldw;

    v8f acc00 = {}, acc01 = {}, acc10 = {}, acc11 = {};
    for (int k = 0; k < K; k += 32) {
        union { v16bf v; v8bf h[2]; } a0, a1, b0, b1;
        a0.h[0] = *(const v8bf*)(Ap0 + k);        // K = abase+0..7
        a0.h[1] = *(const v8bf*)(Ap0 + k + 16);   // K = abase+16..23
        a1.h[0] = *(const v8bf*)(Ap1 + k);
        a1.h[1] = *(const v8bf*)(Ap1 + k + 16);
        b0.h[0] = *(const v8bf*)(Wp0 + k);        // K = bbase+0..7
        b0.h[1] = *(const v8bf*)(Wp0 + k + 16);   // K = bbase+16..23
        b1.h[0] = *(const v8bf*)(Wp1 + k);
        b1.h[1] = *(const v8bf*)(Wp1 + k + 16);
        __builtin_prefetch(Ap0 + k + 64, 0, 1);   // global_prefetch_b8
        __builtin_prefetch(Ap1 + k + 64, 0, 1);
        __builtin_prefetch(Wp0 + k + 64, 0, 1);
        __builtin_prefetch(Wp1 + k + 64, 0, 1);
        acc00 = __builtin_amdgcn_wmma_f32_16x16x32_bf16(
                    false, a0.v, false, b0.v, (short)0, acc00, false, false);
        acc01 = __builtin_amdgcn_wmma_f32_16x16x32_bf16(
                    false, a0.v, false, b1.v, (short)0, acc01, false, false);
        acc10 = __builtin_amdgcn_wmma_f32_16x16x32_bf16(
                    false, a1.v, false, b0.v, (short)0, acc10, false, false);
        acc11 = __builtin_amdgcn_wmma_f32_16x16x32_bf16(
                    false, a1.v, false, b1.v, (short)0, acc11, false, false);
    }

    // C/D layout per 16x16 sub-tile: lane&15 = N, VGPR r = M row (lanes>=16: M+8)
    const int n0 = (nt << 5) + (lane & 15);
    const int m0 = (mt << 5) + ((lane >> 4) << 3);
    float bsA = 0.f, bsB = 0.f;                  // bias for n0 and n0+16
    if (bias1) { bsA += bias1[n0]; bsB += bias1[n0 + 16]; }
    if (bias2) { bsA += bias2[n0]; bsB += bias2[n0 + 16]; }

#pragma unroll
    for (int sub = 0; sub < 4; ++sub) {
        const v8f& acc = (sub == 0) ? acc00 : (sub == 1) ? acc01
                       : (sub == 2) ? acc10 : acc11;
        const int n = n0 + ((sub & 1) << 4);
        const int mb = m0 + ((sub >> 1) << 4);
        const float bs = (sub & 1) ? bsB : bsA;
#pragma unroll
        for (int r = 0; r < 8; ++r) {
            const int m = mb + r;
            float v = acc[r] + bs;
            if (addend) v += addend[(size_t)m * ldadd + n];
            const size_t orow = permute ? ((size_t)(m % pb) * pt + (m / pb))
                                        : (size_t)m;
            C[orow * (size_t)ldc + n] = v;
        }
    }
}

// ---------------------------------------------------------------------------
// Elementwise / gather / pointwise kernels
// ---------------------------------------------------------------------------
__global__ void cvt_f32_bf16(const float* __restrict__ in, __bf16* __restrict__ out, int n)
{
    int i = blockIdx.x * 256 + threadIdx.x;
    if (i < n) out[i] = (__bf16)in[i];
}

__global__ void cvt_rows_bf16(const float* __restrict__ in, int in_ld,
                              __bf16* __restrict__ out, int out_ld, int cols)
{
    int r = blockIdx.x;
    for (int c = threadIdx.x; c < cols; c += blockDim.x)
        out[(size_t)r * out_ld + c] = (__bf16)in[(size_t)r * in_ld + c];
}

__global__ void fill_zero_f32(float* p, int n)
{
    int i = blockIdx.x * 256 + threadIdx.x;
    if (i < n) p[i] = 0.f;
}
__global__ void fill_zero_bf16(__bf16* p, int n)
{
    int i = blockIdx.x * 256 + threadIdx.x;
    if (i < n) p[i] = (__bf16)0.f;
}

// encoder embedding gather: one block per (b,s) token, 256 cols
__global__ void embed_enc(const int* __restrict__ src, const float* __restrict__ emb,
                          __bf16* __restrict__ out)
{
    int bs = blockIdx.x;
    int tok = src[bs];
    out[(size_t)bs * ENC_EMB + threadIdx.x] =
        (__bf16)emb[(size_t)tok * ENC_EMB + threadIdx.x];
}

// decoder embedding gather with SOS shift: dec_in[t] = (t==0) ? SOS : trg[t-1]
__global__ void embed_dec(const int* __restrict__ trg, int t,
                          const float* __restrict__ emb, __bf16* __restrict__ out)
{
    int b = blockIdx.x;
    int tok = (t == 0) ? 1 : trg[b * TRGLEN + (t - 1)];
    out[(size_t)b * DEC_EMB + threadIdx.x] =
        (__bf16)emb[(size_t)tok * DEC_EMB + threadIdx.x];
}

__device__ __forceinline__ float sigm(float x) { return 1.f / (1.f + __expf(-x)); }

// encoder LSTM pointwise (H=256): gate order i,f,g,o
__global__ void lstm_point_enc(const float* __restrict__ gates,
                               float* __restrict__ cstate,
                               __bf16* __restrict__ hstate_bf,
                               float* __restrict__ out_seq,       // pre-offset (t, dir)
                               __bf16* __restrict__ out_seq_bf,   // pre-offset (t, dir)
                               int seq_stride,
                               __bf16* __restrict__ hfin_bf,
                               __bf16* __restrict__ cfin_bf, int fin_off)
{
    const int H = ENC_H;
    int b = blockIdx.x, j = threadIdx.x;
    float i = gates[(size_t)b * 4 * H + j];
    float f = gates[(size_t)b * 4 * H + H + j];
    float g = gates[(size_t)b * 4 * H + 2 * H + j];
    float o = gates[(size_t)b * 4 * H + 3 * H + j];
    float c = cstate[b * H + j];
    c = sigm(f) * c + sigm(i) * tanhf(g);
    float h = sigm(o) * tanhf(c);
    cstate[b * H + j]    = c;
    hstate_bf[b * H + j] = (__bf16)h;
    out_seq  [(size_t)b * seq_stride + j] = h;
    out_seq_bf[(size_t)b * seq_stride + j] = (__bf16)h;
    hfin_bf[b * 512 + fin_off + j] = (__bf16)h;   // overwritten each t; last = final
    cfin_bf[b * 512 + fin_off + j] = (__bf16)c;
}

// decoder LSTM pointwise (H=512): writes h (bf16) into hcat[:, 0:512]
__global__ void lstm_point_dec(const float* __restrict__ gates,
                               float* __restrict__ cstate,
                               __bf16* __restrict__ hcat_bf)
{
    const int H = DEC_H;
    int b = blockIdx.x, j = threadIdx.x;
    float i = gates[(size_t)b * 4 * H + j];
    float f = gates[(size_t)b * 4 * H + H + j];
    float g = gates[(size_t)b * 4 * H + 2 * H + j];
    float o = gates[(size_t)b * 4 * H + 3 * H + j];
    float c = cstate[b * H + j];
    c = sigm(f) * c + sigm(i) * tanhf(g);
    float h = sigm(o) * tanhf(c);
    cstate[b * H + j] = c;
    hcat_bf[(size_t)b * 1024 + j] = (__bf16)h;
}

// Bahdanau attention for one decode step. One block per batch row.
// e[s] = v . tanh(ctx_proj[b,s,:] + q[b,:]) ; softmax over s ;
// context = sum_s score*ctx[b,s,:]  -> bf16 into hcat[:, 512:1024];
// score -> d_out scores region at [b, t, s].
__global__ void __launch_bounds__(256)
attn_step(const float* __restrict__ q, const float* __restrict__ ctx_proj,
          const float* __restrict__ vvec, const float* __restrict__ ctx,
          float* __restrict__ scores_out, int t,
          __bf16* __restrict__ hcat_bf)
{
    __shared__ float qs[ATT];
    __shared__ float es[64];
    __shared__ float smsum;
    const int b = blockIdx.x;
    for (int d = threadIdx.x; d < ATT; d += 256) qs[d] = q[(size_t)b * ATT + d];
    __syncthreads();

    const int wave = threadIdx.x >> 5, lane = threadIdx.x & 31;
    for (int s = wave; s < SRCLEN; s += 8) {
        const float* cp = ctx_proj + ((size_t)b * SRCLEN + s) * ATT;
        float acc = 0.f;
        for (int d = lane; d < ATT; d += 32)
            acc += vvec[d] * tanhf(cp[d] + qs[d]);
        for (int off = 16; off; off >>= 1) acc += __shfl_xor(acc, off, 32);
        if (lane == 0) es[s] = acc;
    }
    __syncthreads();
    if (threadIdx.x == 0) {
        float m = -3.0e38f;
        for (int s = 0; s < SRCLEN; ++s) m = fmaxf(m, es[s]);
        float ss = 0.f;
        for (int s = 0; s < SRCLEN; ++s) { es[s] = __expf(es[s] - m); ss += es[s]; }
        smsum = ss;
    }
    __syncthreads();
    const float inv = 1.f / smsum;
    for (int s = threadIdx.x; s < SRCLEN; s += 256)
        scores_out[((size_t)b * TRGLEN + t) * SRCLEN + s] = es[s] * inv;
    for (int d = threadIdx.x; d < DEC_H; d += 256) {
        float acc = 0.f;
        for (int s = 0; s < SRCLEN; ++s)
            acc += es[s] * inv * ctx[((size_t)b * SRCLEN + s) * DEC_H + d];
        hcat_bf[(size_t)b * 1024 + 512 + d] = (__bf16)acc;
    }
}

// ---------------------------------------------------------------------------
// Host orchestration
// ---------------------------------------------------------------------------
static inline void launch_gemm(hipStream_t st,
                               const __bf16* A, int lda, const __bf16* W, int ldw,
                               float* C, int ldc,
                               const float* b1, const float* b2,
                               const float* add, int ldadd,
                               int M, int N, int K,
                               int permute = 0, int pb = 1, int pt = 1)
{
    int tiles  = (M / 32) * (N / 32);     // 32x32 macro-tiles, one wave each
    int blocks = (tiles + 7) / 8;
    gemm_bf16_wmma<<<blocks, 256, 0, st>>>(A, lda, W, ldw, C, ldc, b1, b2,
                                           add, ldadd, M, N, K, permute, pb, pt);
}

static inline void launch_cvt(hipStream_t st, const float* in, __bf16* out, int n)
{
    cvt_f32_bf16<<<(n + 255) / 256, 256, 0, st>>>(in, out, n);
}

extern "C" void kernel_launch(void* const* d_in, const int* in_sizes, int n_in,
                              void* d_out, int out_size, void* d_ws, size_t ws_size,
                              hipStream_t stream)
{
    (void)in_sizes; (void)n_in; (void)out_size; (void)ws_size;
    // ---- inputs (setup_inputs dict order, params flattened in insertion order)
    const int*   src    = (const int*)d_in[0];
    const int*   trg    = (const int*)d_in[1];
    const float* encemb = (const float*)d_in[2];
    const float* Wih1f  = (const float*)d_in[3];
    const float* Whh1f  = (const float*)d_in[4];
    const float* bih1f  = (const float*)d_in[5];
    const float* bhh1f  = (const float*)d_in[6];
    const float* Wih1b  = (const float*)d_in[7];
    const float* Whh1b  = (const float*)d_in[8];
    const float* bih1b  = (const float*)d_in[9];
    const float* bhh1b  = (const float*)d_in[10];
    const float* Wih2f  = (const float*)d_in[11];
    const float* Whh2f  = (const float*)d_in[12];
    const float* bih2f  = (const float*)d_in[13];
    const float* bhh2f  = (const float*)d_in[14];
    const float* Wih2b  = (const float*)d_in[15];
    const float* Whh2b  = (const float*)d_in[16];
    const float* bih2b  = (const float*)d_in[17];
    const float* bhh2b  = (const float*)d_in[18];
    const float* W_ed   = (const float*)d_in[19];
    const float* b_ed   = (const float*)d_in[20];
    const float* W_ctx  = (const float*)d_in[21];
    const float* b_ctx  = (const float*)d_in[22];
    const float* W_q    = (const float*)d_in[23];
    const float* b_q    = (const float*)d_in[24];
    const float* vvec   = (const float*)d_in[25];
    const float* decemb = (const float*)d_in[26];
    const float* Wih_d  = (const float*)d_in[27];
    const float* Whh_d  = (const float*)d_in[28];
    const float* bih_d  = (const float*)d_in[29];
    const float* bhh_d  = (const float*)d_in[30];
    const float* W_cat  = (const float*)d_in[31];
    const float* b_cat  = (const float*)d_in[32];
    const float* W_out  = (const float*)d_in[33];
    const float* b_out  = (const float*)d_in[34];

    float* out_logits = (float*)d_out;                       // [B,T,V]
    float* out_scores = (float*)d_out + (size_t)BATCH * TRGLEN * VOCAB;  // [B,T,S]

    // ---- workspace bump allocator
    char* wsp = (char*)d_ws;
    auto alloc = [&](size_t bytes) -> char* {
        char* r = wsp; wsp += (bytes + 255) & ~(size_t)255; return r;
    };
    auto fbuf = [&](size_t n) { return (float*)alloc(n * 4); };
    auto bbuf = [&](size_t n) { return (__bf16*)alloc(n * 2); };

    __bf16* x_bf     = bbuf((size_t)BS * ENC_EMB);
    __bf16* wih1f_bf = bbuf(1024 * 256);  __bf16* whh1f_bf = bbuf(1024 * 256);
    __bf16* wih1b_bf = bbuf(1024 * 256);  __bf16* whh1b_bf = bbuf(1024 * 256);
    __bf16* wih2f_bf = bbuf(1024 * 512);  __bf16* whh2f_bf = bbuf(1024 * 256);
    __bf16* wih2b_bf = bbuf(1024 * 512);  __bf16* whh2b_bf = bbuf(1024 * 256);
    __bf16* wed_bf   = bbuf(512 * 512);
    __bf16* wctx_bf  = bbuf(512 * 512);
    __bf16* wq_bf    = bbuf(512 * 512);
    __bf16* wihd_bf  = bbuf(2048 * 256);  __bf16* whhd_bf = bbuf(2048 * 512);
    __bf16* wcat_bf  = bbuf(256 * 1024);
    __bf16* wout_bf  = bbuf((size_t)VOCAB * 256);

    float*  pre_f    = fbuf((size_t)BS * 1024);
    float*  pre_b    = fbuf((size_t)BS * 1024);
    float*  out1     = fbuf((size_t)BS * 512);
    __bf16* out1_bf  = bbuf((size_t)BS * 512);
    float*  out2     = fbuf((size_t)BS * 512);
    __bf16* out2_bf  = bbuf((size_t)BS * 512);
    float*  ctx      = fbuf((size_t)BS * 512);
    __bf16* ctx_bf   = bbuf((size_t)BS * 512);
    float*  ctxproj  = fbuf((size_t)BS * 512);
    __bf16* hstate   = bbuf(BATCH * ENC_H);
    float*  cst_enc  = fbuf(BATCH * ENC_H);
    __bf16* hfin_bf  = bbuf(BATCH * 512);
    __bf16* cfin_bf  = bbuf(BATCH * 512);
    float*  h0       = fbuf(BATCH * DEC_H);
    float*  cst_dec  = fbuf(BATCH * DEC_H);
    __bf16* hcat_bf  = bbuf(BATCH * 1024);     // [h (512) | context (512)]
    float*  gat_enc  = fbuf(BATCH * 1024);
    float*  gat_dec  = fbuf(BATCH * 2048);
    __bf16* embd_bf  = bbuf(BATCH * DEC_EMB);
    float*  qbuf     = fbuf(BATCH * ATT);
    float*  catf     = fbuf(BATCH * 256);
    __bf16* catall   = bbuf((size_t)BT * 256);

    // ---- weight conversions (fp32 -> bf16, row-major [N,K] preserved)
    launch_cvt(stream, Wih1f, wih1f_bf, 1024 * 256);
    launch_cvt(stream, Whh1f, whh1f_bf, 1024 * 256);
    launch_cvt(stream, Wih1b, wih1b_bf, 1024 * 256);
    launch_cvt(stream, Whh1b, whh1b_bf, 1024 * 256);
    launch_cvt(stream, Wih2f, wih2f_bf, 1024 * 512);
    launch_cvt(stream, Whh2f, whh2f_bf, 1024 * 256);
    launch_cvt(stream, Wih2b, wih2b_bf, 1024 * 512);
    launch_cvt(stream, Whh2b, whh2b_bf, 1024 * 256);
    launch_cvt(stream, W_ed,  wed_bf,   512 * 512);
    launch_cvt(stream, W_ctx, wctx_bf,  512 * 512);
    launch_cvt(stream, W_q,   wq_bf,    512 * 512);
    launch_cvt(stream, Wih_d, wihd_bf,  2048 * 256);
    launch_cvt(stream, Whh_d, whhd_bf,  2048 * 512);
    launch_cvt(stream, W_cat, wcat_bf,  256 * 1024);
    launch_cvt(stream, W_out, wout_bf,  VOCAB * 256);

    // ---- encoder embedding
    embed_enc<<<BS, 256, 0, stream>>>(src, encemb, x_bf);

    // ---- BiLSTM layer runner
    auto run_bilstm = [&](const __bf16* in_bf, int in_dim,
                          const __bf16* wihf, const __bf16* whhf,
                          const float* bihf, const float* bhhf,
                          const __bf16* wihb, const __bf16* whhb,
                          const float* bihb, const float* bhhb,
                          float* out_seq, __bf16* out_seq_bf) {
        // full-sequence input projections: [BS,in] @ Wih^T -> [BS,1024]
        launch_gemm(stream, in_bf, in_dim, wihf, in_dim, pre_f, 1024,
                    bihf, bhhf, nullptr, 0, BS, 1024, in_dim);
        launch_gemm(stream, in_bf, in_dim, wihb, in_dim, pre_b, 1024,
                    bihb, bhhb, nullptr, 0, BS, 1024, in_dim);
        for (int dir = 0; dir < 2; ++dir) {
            const __bf16* whh = dir ? whhb : whhf;
            const float*  pre = dir ? pre_b : pre_f;
            fill_zero_bf16<<<(BATCH * ENC_H + 255) / 256, 256, 0, stream>>>(hstate, BATCH * ENC_H);
            fill_zero_f32 <<<(BATCH * ENC_H + 255) / 256, 256, 0, stream>>>(cst_enc, BATCH * ENC_H);
            for (int step = 0; step < SRCLEN; ++step) {
                int t = dir ? (SRCLEN - 1 - step) : step;
                // gates = h @ Whh^T + pre[:, t, :]
                launch_gemm(stream, hstate, ENC_H, whh, ENC_H, gat_enc, 1024,
                            nullptr, nullptr, pre + (size_t)t * 1024, SRCLEN * 1024,
                            BATCH, 1024, ENC_H);
                lstm_point_enc<<<BATCH, ENC_H, 0, stream>>>(
                    gat_enc, cst_enc, hstate,
                    out_seq    + (size_t)t * 512 + dir * ENC_H,
                    out_seq_bf + (size_t)t * 512 + dir * ENC_H,
                    SRCLEN * 512,
                    hfin_bf, cfin_bf, dir * ENC_H);
            }
        }
    };

    run_bilstm(x_bf, ENC_EMB, wih1f_bf, whh1f_bf, bih1f, bhh1f,
               wih1b_bf, whh1b_bf, bih1b, bhh1b, out1, out1_bf);
    run_bilstm(out1_bf, 2 * ENC_H, wih2f_bf, whh2f_bf, bih2f, bhh2f,
               wih2b_bf, whh2b_bf, bih2b, bhh2b, out2, out2_bf);

    // ---- encoder-to-decoder projections
    // ctx = out2 @ W_ed^T + b_ed   [BS,512]
    launch_gemm(stream, out2_bf, 512, wed_bf, 512, ctx, 512,
                b_ed, nullptr, nullptr, 0, BS, 512, 512);
    launch_cvt(stream, ctx, ctx_bf, BS * 512);
    // ctx_proj = ctx @ W_ctx^T + b_ctx   [BS,512]
    launch_gemm(stream, ctx_bf, 512, wctx_bf, 512, ctxproj, 512,
                b_ctx, nullptr, nullptr, 0, BS, 512, 512);
    // h0 = concat(hf,hb) @ W_ed^T + b_ed ;  c0 likewise (straight into c-state)
    launch_gemm(stream, hfin_bf, 512, wed_bf, 512, h0, 512,
                b_ed, nullptr, nullptr, 0, BATCH, 512, 512);
    launch_gemm(stream, cfin_bf, 512, wed_bf, 512, cst_dec, 512,
                b_ed, nullptr, nullptr, 0, BATCH, 512, 512);
    cvt_rows_bf16<<<BATCH, 256, 0, stream>>>(h0, 512, hcat_bf, 1024, 512);

    // ---- decoder (teacher forcing); W_out projection hoisted out of the loop
    for (int t = 0; t < TRGLEN; ++t) {
        embed_dec<<<BATCH, DEC_EMB, 0, stream>>>(trg, t, decemb, embd_bf);
        // gates = emb @ Wih_d^T + (bih_d + bhh_d)
        launch_gemm(stream, embd_bf, DEC_EMB, wihd_bf, DEC_EMB, gat_dec, 2048,
                    bih_d, bhh_d, nullptr, 0, BATCH, 2048, DEC_EMB);
        // gates += h @ Whh_d^T   (h is hcat[:, 0:512], lda = 1024)
        launch_gemm(stream, hcat_bf, 1024, whhd_bf, DEC_H, gat_dec, 2048,
                    nullptr, nullptr, gat_dec, 2048, BATCH, 2048, DEC_H);
        lstm_point_dec<<<BATCH, DEC_H, 0, stream>>>(gat_dec, cst_dec, hcat_bf);
        // q = h @ W_q^T + b_q
        launch_gemm(stream, hcat_bf, 1024, wq_bf, DEC_H, qbuf, ATT,
                    b_q, nullptr, nullptr, 0, BATCH, ATT, DEC_H);
        attn_step<<<BATCH, 256, 0, stream>>>(qbuf, ctxproj, vvec, ctx,
                                             out_scores, t, hcat_bf);
        // cat = [h | context] @ W_cat^T + b_cat   [64,256]
        launch_gemm(stream, hcat_bf, 1024, wcat_bf, 1024, catf, 256,
                    b_cat, nullptr, nullptr, 0, BATCH, 256, 1024);
        cvt_rows_bf16<<<BATCH, 256, 0, stream>>>(catf, 256,
                                                 catall + (size_t)t * BATCH * 256, 256, 256);
    }

    // ---- batched output projection: [T*B,256] @ W_out^T + b_out -> d_out[B,T,V]
    // row r = t*B+b  stored at output row b*T+t  (permute: pb=BATCH, pt=TRGLEN)
    launch_gemm(stream, catall, 256, wout_bf, 256, out_logits, VOCAB,
                b_out, nullptr, nullptr, 0, BT, VOCAB, 256,
                /*permute=*/1, /*pb=*/BATCH, /*pt=*/TRGLEN);
}
